// SparseConv3d_10239202034089
// MI455X (gfx1250) — matrix operand order repaired and checked by reading, not compiled, not run on hardware
//
#include <hip/hip_runtime.h>
#include <hip/hip_bf16.h>

#define GRID_  256
#define PAD_   1
#define G_     (GRID_ + 2 * PAD_)     // 258
#define CIN    32
#define COUT   32
#define KV     27
#define HBITS  20
#define HSIZE  (1u << HBITS)          // 1,048,576 slots
#define HMASK  (HSIZE - 1u)
#define TM     32                     // rows (points) per wave
#define WF_ELEMS (KV * 2 * 32 * 16)   // bf16 elements in fragment store (27648)

typedef __attribute__((ext_vector_type(16))) __bf16 v16bf;
typedef __attribute__((ext_vector_type(8)))  float  v8f;

__device__ __forceinline__ unsigned short f2bf(float f) {
    // round-to-nearest-even f32 -> bf16
    unsigned u = __float_as_uint(f);
    unsigned r = ((u >> 16) & 1u) + 0x7FFFu;
    return (unsigned short)((u + r) >> 16);
}

// ---------------- hash table: int2 slots (key, val), one b64 per probe ----------------

__global__ void hash_init_kernel(int2* __restrict__ table) {
    unsigned i = blockIdx.x * blockDim.x + threadIdx.x;
    if (i < HSIZE) table[i] = make_int2(-1, 0);
}

__global__ void hash_insert_kernel(const int* __restrict__ coords, int n,
                                   int2* __restrict__ table) {
    int i = blockIdx.x * blockDim.x + threadIdx.x;
    if (i >= n) return;
    int b = coords[i * 4 + 0];
    int x = coords[i * 4 + 1] + PAD_;
    int y = coords[i * 4 + 2] + PAD_;
    int z = coords[i * 4 + 3] + PAD_;
    int key = ((b * G_ + x) * G_ + y) * G_ + z;   // < 34.4M, fits int32
    unsigned h = ((unsigned)key * 2654435761u) & HMASK;
    for (;;) {
        int prev = atomicCAS(&table[h].x, -1, key);
        if (prev == -1 || prev == key) { table[h].y = i; break; }
        h = (h + 1) & HMASK;
    }
}

__device__ __forceinline__ int hash_lookup(const int2* __restrict__ t, int qkey) {
    unsigned h = ((unsigned)qkey * 2654435761u) & HMASK;
    int2 kv = t[h];                    // single global_load_b64, L2-resident
    while (kv.x != qkey) {
        if (kv.x == -1) return -1;
        h = (h + 1) & HMASK;
        kv = t[h];
    }
    return kv.y;
}

// ---------------- weight -> bf16 B-matrix fragments ----------------
// wf[((k*2 + h)*32 + lane)*16 + j] = bf16( weight[k][Kbase + (j<8 ? j : j+8)][(lane&15) + 16*h] )
// Kbase = 0 for lanes 0-15, 8 for lanes 16-31 (matches A fragment K order).
__global__ void build_wfrag_kernel(const float* __restrict__ w, unsigned short* __restrict__ wf) {
    int t = blockIdx.x * blockDim.x + threadIdx.x;
    if (t >= WF_ELEMS) return;
    int j    = t & 15;
    int lane = (t >> 4) & 31;
    int h    = (t >> 9) & 1;
    int k    = t >> 10;
    int c    = (lane & 15) + 16 * h;
    int K    = ((lane < 16) ? 0 : 8) + ((j < 8) ? j : j + 8);
    wf[t] = f2bf(w[(k * CIN + K) * COUT + c]);
}

// ---------------- async global -> LDS b128 copy (CDNA5 async path) ----------------
__device__ __forceinline__ void async_copy_b128_to_lds(void* lds_dst, const void* gsrc) {
    // Generic shared-pointer low 32 bits == LDS byte offset (flat->LDS truncation rule).
    unsigned ldsoff = (unsigned)(size_t)lds_dst;
    unsigned long long ga = (unsigned long long)(size_t)gsrc;
    asm volatile("global_load_async_to_lds_b128 %0, %1, off"
                 :: "v"(ldsoff), "v"(ga)
                 : "memory");
}

__device__ __forceinline__ void wait_asynccnt0() {
    asm volatile("s_wait_asynccnt 0" ::: "memory");
}

// ---------------- A fragment gather ----------------
__device__ __forceinline__ v16bf gather_a(const float* __restrict__ feats, int idx, int colbase) {
    union { v16bf v; unsigned short u[16]; } A;
    if (idx >= 0) {
        const float4* fp = (const float4*)(feats + (long)idx * CIN);
        float4 a0 = fp[(colbase >> 2) + 0];
        float4 a1 = fp[(colbase >> 2) + 1];
        float4 a2 = fp[(colbase >> 2) + 4];
        float4 a3 = fp[(colbase >> 2) + 5];
        A.u[0]  = f2bf(a0.x); A.u[1]  = f2bf(a0.y); A.u[2]  = f2bf(a0.z); A.u[3]  = f2bf(a0.w);
        A.u[4]  = f2bf(a1.x); A.u[5]  = f2bf(a1.y); A.u[6]  = f2bf(a1.z); A.u[7]  = f2bf(a1.w);
        A.u[8]  = f2bf(a2.x); A.u[9]  = f2bf(a2.y); A.u[10] = f2bf(a2.z); A.u[11] = f2bf(a2.w);
        A.u[12] = f2bf(a3.x); A.u[13] = f2bf(a3.y); A.u[14] = f2bf(a3.z); A.u[15] = f2bf(a3.w);
    } else {
        #pragma unroll
        for (int j = 0; j < 16; ++j) A.u[j] = 0;
    }
    return A.v;
}

// ---------------- main conv: one wave32 per 32-point tile ----------------
__global__ __launch_bounds__(256)
void conv_wmma_kernel(const int*   __restrict__ coords,
                      const float* __restrict__ feats,
                      const float* __restrict__ bias,
                      const int*   __restrict__ offsets,
                      const int2*  __restrict__ table,
                      const unsigned short* __restrict__ wfrag,
                      float* __restrict__ out, int n)
{
    __shared__ unsigned short swf[WF_ELEMS];    // 55,296 bytes of 320 KB/WGP

    // stage all 27x2 B fragments into LDS via the async copy engine
    {
        uint4*       dst = (uint4*)swf;
        const uint4* src = (const uint4*)wfrag;
        for (int i = threadIdx.x; i < WF_ELEMS / 8; i += blockDim.x)
            async_copy_b128_to_lds(&dst[i], &src[i]);
        wait_asynccnt0();
    }
    __syncthreads();

    const int lane  = threadIdx.x & 31;
    const int wid   = (int)((blockIdx.x * blockDim.x + threadIdx.x) >> 5);
    const int tiles = (n + TM - 1) / TM;
    if (wid >= tiles) return;

    // Lanes L and L+16 both handle rows (L&15) and 16+(L&15); they cover different K halves.
    int r0 = wid * TM + (lane & 15);
    int r1 = r0 + 16;
    if (r0 >= n) r0 = n - 1;                    // tail clamp (stores guarded below)
    if (r1 >= n) r1 = n - 1;

    // base voxel keys for both rows
    int bk0, bk1;
    {
        int b = coords[r0 * 4 + 0], x = coords[r0 * 4 + 1] + PAD_;
        int y = coords[r0 * 4 + 2] + PAD_, z = coords[r0 * 4 + 3] + PAD_;
        bk0 = ((b * G_ + x) * G_ + y) * G_ + z;
        b = coords[r1 * 4 + 0]; x = coords[r1 * 4 + 1] + PAD_;
        y = coords[r1 * 4 + 2] + PAD_; z = coords[r1 * 4 + 3] + PAD_;
        bk1 = ((b * G_ + x) * G_ + y) * G_ + z;
    }
    const int colbase = (lane < 16) ? 0 : 8;    // A fragment K columns {cb..cb+7, cb+16..cb+23}

    // warm L2/probe pipeline: prefetch the first-probe hash slot of all 27 offsets
    #pragma unroll
    for (int k = 0; k < KV; ++k) {
        const int ox = offsets[k * 4 + 1];
        const int oy = offsets[k * 4 + 2];
        const int oz = offsets[k * 4 + 3];
        const int dk = (ox * G_ + oy) * G_ + oz;
        unsigned h0 = ((unsigned)(bk0 + dk) * 2654435761u) & HMASK;
        unsigned h1 = ((unsigned)(bk1 + dk) * 2654435761u) & HMASK;
        __builtin_prefetch(&table[h0], 0, 1);   // global_prefetch_b8
        __builtin_prefetch(&table[h1], 0, 1);
    }

    v8f acc00 = {}, acc01 = {};                 // rows 0..15  x cols {0..15, 16..31}
    v8f acc10 = {}, acc11 = {};                 // rows 16..31 x cols {0..15, 16..31}

    const uint4* sB = (const uint4*)swf;

    #pragma unroll 3
    for (int k = 0; k < KV; ++k) {
        // wave-uniform offset -> scalar loads + SALU key delta
        const int ox = offsets[k * 4 + 1];
        const int oy = offsets[k * 4 + 2];
        const int oz = offsets[k * 4 + 3];
        const int dk = (ox * G_ + oy) * G_ + oz;

        const int idx0 = hash_lookup(table, bk0 + dk);
        const int idx1 = hash_lookup(table, bk1 + dk);

        v16bf A0 = gather_a(feats, idx0, colbase);
        v16bf A1 = gather_a(feats, idx1, colbase);

        union BU { v16bf v; uint4 q[2]; } B0, B1;
        const int fb0 = ((k * 2 + 0) * 32 + lane) * 2;   // uint4 units
        const int fb1 = ((k * 2 + 1) * 32 + lane) * 2;
        B0.q[0] = sB[fb0]; B0.q[1] = sB[fb0 + 1];        // ds_load_b128 x2
        B1.q[0] = sB[fb1]; B1.q[1] = sB[fb1 + 1];

        acc00 = __builtin_amdgcn_wmma_f32_16x16x32_bf16(false, A0, false, B0.v,
                                                        (short)0, acc00, false, false);
        acc01 = __builtin_amdgcn_wmma_f32_16x16x32_bf16(false, A0, false, B1.v,
                                                        (short)0, acc01, false, false);
        acc10 = __builtin_amdgcn_wmma_f32_16x16x32_bf16(false, A1, false, B0.v,
                                                        (short)0, acc10, false, false);
        acc11 = __builtin_amdgcn_wmma_f32_16x16x32_bf16(false, A1, false, B1.v,
                                                        (short)0, acc11, false, false);
    }

    // C/D layout: lanes 0-15 -> N=lane, M=0..7 in v[0..7]; lanes 16-31 -> N=lane-16, M=8..15
    const int   c0 = lane & 15;
    const float b0 = bias[c0];
    const float b1 = bias[c0 + 16];
    const int   mb0 = wid * TM + ((lane < 16) ? 0 : 8);        // A-tile 0
    const int   mb1 = mb0 + 16;                                 // A-tile 1
    #pragma unroll
    for (int v = 0; v < 8; ++v) {
        int m = mb0 + v;
        if (m < n) {
            out[(long)m * COUT + c0]      = acc00[v] + b0;
            out[(long)m * COUT + c0 + 16] = acc01[v] + b1;
        }
        m = mb1 + v;
        if (m < n) {
            out[(long)m * COUT + c0]      = acc10[v] + b0;
            out[(long)m * COUT + c0 + 16] = acc11[v] + b1;
        }
    }
}

extern "C" void kernel_launch(void* const* d_in, const int* in_sizes, int n_in,
                              void* d_out, int out_size, void* d_ws, size_t ws_size,
                              hipStream_t stream) {
    const int*   coords  = (const int*)  d_in[0];
    const float* feats   = (const float*)d_in[1];
    const float* weight  = (const float*)d_in[2];
    const float* bias    = (const float*)d_in[3];
    const int*   offsets = (const int*)  d_in[4];
    const int n = in_sizes[0] / 4;               // coords is N x 4

    char* ws = (char*)d_ws;
    int2* table = (int2*)ws;                                        // 8 MB
    unsigned short* wfrag = (unsigned short*)(ws + (size_t)HSIZE * sizeof(int2));

    hash_init_kernel<<<(HSIZE + 255) / 256, 256, 0, stream>>>(table);
    hash_insert_kernel<<<(n + 255) / 256, 256, 0, stream>>>(coords, n, table);
    build_wfrag_kernel<<<(WF_ELEMS + 255) / 256, 256, 0, stream>>>(weight, wfrag);

    const int  tiles   = (n + TM - 1) / TM;      // one wave32 per 32-point tile
    const long threads = (long)tiles * 32;
    conv_wmma_kernel<<<(int)((threads + 255) / 256), 256, 0, stream>>>(
        coords, feats, bias, offsets, table, wfrag, (float*)d_out, n);
}
